// MultiHeadAttention_41111426957888
// MI455X (gfx1250) — matrix operand order repaired and checked by reading, not compiled, not run on hardware
//
#include <hip/hip_runtime.h>

// ---------------------------------------------------------------------------
// Fused additive-attention layer for MI455X (gfx1250, wave32, WMMA).
// One workgroup (256 threads = 8 waves) per batch row; one wave per head.
// Projections k@Wk + p@Wp (fused accumulation) and v@Wv run on
// v_wmma_f32_16x16x32_bf16; agg_v reduction, softmax, ctx and the tiny
// 1x256 GEMMs (q-proj / out-proj) are fused epilogues.
// ---------------------------------------------------------------------------

typedef __bf16 bf16_t;
typedef __attribute__((ext_vector_type(16))) __bf16 v16bf;
typedef __attribute__((ext_vector_type(8)))  float  v8f;

#define BDIM   2048
#define NDIM   128
#define DDIM   256
#define HDIM   8
#define APITCH 264   // bf16 elems/row: 528B = 132 dwords; 132%64=4 -> conflict-free, 16B aligned

__device__ __forceinline__ v8f zero8() {
    v8f r;
    #pragma unroll
    for (int i = 0; i < 8; ++i) r[i] = 0.0f;
    return r;
}

// B fragment (32x16, bf16) from transposed weights WT[n][k] in global (L2-resident).
// Lane layout: col = nBase + (lane&15); lanes 0-15 hold K kBase..kBase+15,
// lanes 16-31 hold K kBase+16..kBase+31 -> 32 contiguous bytes per lane.
__device__ __forceinline__ v16bf load_bfrag(const bf16_t* __restrict__ Wt,
                                            int nBase, int kBase, int lane) {
    int row = nBase + (lane & 15);
    int kk  = kBase + ((lane & 16) ? 16 : 0);
    const uint4* p = reinterpret_cast<const uint4*>(Wt + row * DDIM + kk);
    union { uint4 u[2]; v16bf v; } t;
    t.u[0] = p[0];
    t.u[1] = p[1];
    return t.v;
}

// A fragment (16x32, bf16) from LDS row-major tile.
// Lane 0-15: row mBase+lane, K kBase..kBase+15;
// lane 16-31: row mBase+lane-16, K kBase+16..kBase+31.
__device__ __forceinline__ v16bf load_afrag(const bf16_t* base,
                                            int mBase, int kBase, int lane) {
    int row = mBase + (lane & 15);
    int kh  = (lane & 16) ? 16 : 0;
    const bf16_t* p = base + row * APITCH + kBase + kh;
    union { uint4 u[2]; v16bf v; } t;
    t.u[0] = *reinterpret_cast<const uint4*>(p);
    t.u[1] = *reinterpret_cast<const uint4*>(p + 8);
    return t.v;
}

// fp32 [256][256] weights -> bf16 transposed [n][k], 5 matrices packed.
__global__ void prep_weights(const float* __restrict__ wq, const float* __restrict__ wk,
                             const float* __restrict__ wp, const float* __restrict__ wv,
                             const float* __restrict__ wo, bf16_t* __restrict__ dst) {
    int t  = blockIdx.x * blockDim.x + threadIdx.x;   // 0 .. 5*65536-1
    int m  = t >> 16;
    int nk = t & 65535;
    int n  = nk >> 8;
    int kk = nk & 255;
    const float* src = (m == 0) ? wq : (m == 1) ? wk : (m == 2) ? wp : (m == 3) ? wv : wo;
    dst[t] = (bf16_t)src[kk * DDIM + n];
}

__global__ __launch_bounds__(256, 1)
void mha_fused(const float* __restrict__ gq, const float* __restrict__ gk,
               const float* __restrict__ gv, const float* __restrict__ gpw,
               const int*   __restrict__ gmask, const bf16_t* __restrict__ wt,
               const float* __restrict__ wq_b, const float* __restrict__ wk_b,
               const float* __restrict__ wv_b, const float* __restrict__ wp_b,
               const float* __restrict__ out_b, const float* __restrict__ aggv,
               const float* __restrict__ posb, const float* __restrict__ bias_k,
               const float* __restrict__ bias_v, const float* __restrict__ bias_posb,
               const float* __restrict__ bias_pw, float* __restrict__ out) {
    __shared__ bf16_t aBuf[NDIM][APITCH] __attribute__((aligned(16)));
    __shared__ float  qpS[DDIM];
    __shared__ float  sAdd[DDIM];
    __shared__ float  wS[HDIM][132];
    __shared__ float  ctxS[DDIM];

    const int b    = blockIdx.x;
    const int tid  = threadIdx.x;
    const int lane = tid & 31;
    const int wave = tid >> 5;          // == head index
    const int n0   = wave * 32;         // first output column of this head

    const bf16_t* WqT = wt + 0 * DDIM * DDIM;
    const bf16_t* WkT = wt + 1 * DDIM * DDIM;
    const bf16_t* WpT = wt + 2 * DDIM * DDIM;
    const bf16_t* WvT = wt + 3 * DDIM * DDIM;
    const bf16_t* WoT = wt + 4 * DDIM * DDIM;

    // ---- stage q row, compute qp = q@Wq + wq_b (1x256, VALU) ----------------
    ctxS[tid] = gq[b * DDIM + tid];
    __syncthreads();
    {
        float acc = wq_b[tid];
        const bf16_t* wrow = WqT + tid * DDIM;
        #pragma unroll 8
        for (int kk = 0; kk < DDIM; ++kk) acc += ctxS[kk] * (float)wrow[kk];
        qpS[tid]  = acc;
        sAdd[tid] = acc + wk_b[tid] + wp_b[tid];   // column-wise addend of S
    }
    __syncthreads();

    // per-head constant: aggv . sAdd  and raw bias-token logit ---------------
    float headConst, logitBias;
    {
        const int c = n0 + lane;
        float t0 = aggv[c] * sAdd[c];
        float t1 = aggv[c] * (qpS[c] + bias_k[c] + bias_pw[c]);
        #pragma unroll
        for (int off = 16; off > 0; off >>= 1) {
            t0 += __shfl_xor(t0, off, 32);
            t1 += __shfl_xor(t1, off, 32);
        }
        headConst = t0;
        logitBias = t1;
    }

    // cooperative fp32->bf16 tile loader (128x256) ---------------------------
    auto loadA = [&](const float* __restrict__ src) {
        const float4* s4 = reinterpret_cast<const float4*>(src + (size_t)b * NDIM * DDIM);
        #pragma unroll 4
        for (int it = 0; it < 32; ++it) {
            int idx = it * 256 + tid;        // float4 index; 64 per row
            int row = idx >> 6;
            int c4  = (idx & 63) << 2;
            float4 f = s4[idx];
            bf16_t* d = &aBuf[row][c4];
            d[0] = (bf16_t)f.x; d[1] = (bf16_t)f.y;
            d[2] = (bf16_t)f.z; d[3] = (bf16_t)f.w;
        }
    };

    // ---- S = k@Wk + p@Wp (register accumulators, 8 row tiles x 2 col tiles)
    v8f accS[8][2];
    #pragma unroll
    for (int mt = 0; mt < 8; ++mt) { accS[mt][0] = zero8(); accS[mt][1] = zero8(); }

    loadA(gk);
    __syncthreads();
    for (int kk = 0; kk < 8; ++kk) {
        const int kb = kk * 32;
        v16bf b0 = load_bfrag(WkT, n0,      kb, lane);
        v16bf b1 = load_bfrag(WkT, n0 + 16, kb, lane);
        #pragma unroll
        for (int mt = 0; mt < 8; ++mt) {
            v16bf a = load_afrag(&aBuf[0][0], mt * 16, kb, lane);
            accS[mt][0] = __builtin_amdgcn_wmma_f32_16x16x32_bf16(false, a, false, b0,
                              (short)0, accS[mt][0], false, false);
            accS[mt][1] = __builtin_amdgcn_wmma_f32_16x16x32_bf16(false, a, false, b1,
                              (short)0, accS[mt][1], false, false);
        }
    }
    __syncthreads();

    loadA(gpw);
    __syncthreads();
    for (int kk = 0; kk < 8; ++kk) {
        const int kb = kk * 32;
        v16bf b0 = load_bfrag(WpT, n0,      kb, lane);
        v16bf b1 = load_bfrag(WpT, n0 + 16, kb, lane);
        #pragma unroll
        for (int mt = 0; mt < 8; ++mt) {
            v16bf a = load_afrag(&aBuf[0][0], mt * 16, kb, lane);
            accS[mt][0] = __builtin_amdgcn_wmma_f32_16x16x32_bf16(false, a, false, b0,
                              (short)0, accS[mt][0], false, false);
            accS[mt][1] = __builtin_amdgcn_wmma_f32_16x16x32_bf16(false, a, false, b1,
                              (short)0, accS[mt][1], false, false);
        }
    }
    __syncthreads();

    // ---- logits: fold aggv into C-tile epilogue ----------------------------
    // C layout: VGPR r -> row mBase + r (+8 for lanes>=16), col = nBase + (lane&15)
    {
        const float av0 = aggv[n0 + (lane & 15)];
        const float av1 = aggv[n0 + 16 + (lane & 15)];
        #pragma unroll
        for (int mt = 0; mt < 8; ++mt) {
            float s[8];
            #pragma unroll
            for (int r = 0; r < 8; ++r)
                s[r] = accS[mt][0][r] * av0 + accS[mt][1][r] * av1;
            #pragma unroll
            for (int off = 1; off < 16; off <<= 1) {
                #pragma unroll
                for (int r = 0; r < 8; ++r) s[r] += __shfl_xor(s[r], off, 32);
            }
            if ((lane & 15) == 0) {
                int rowBase = mt * 16 + ((lane & 16) ? 8 : 0);
                #pragma unroll
                for (int r = 0; r < 8; ++r) wS[wave][rowBase + r] = headConst + s[r];
            }
        }
    }
    __syncthreads();

    // ---- leaky-relu / scale / pos / mask + softmax over 129 (per wave) -----
    {
        const float kScale = 0.17677669529663687f;     // 1/sqrt(32)
        float L[5];
        #pragma unroll
        for (int i = 0; i < 4; ++i) {
            int n   = lane + i * 32;
            float l = wS[wave][n];
            l = (l > 0.0f) ? l : 0.2f * l;
            l = l * kScale + posb[n] + (float)gmask[b * NDIM + n] * -1.0e12f;
            L[i] = l;
        }
        {
            float l = logitBias;
            l = (l > 0.0f) ? l : 0.2f * l;
            l = l * kScale + bias_posb[0];
            L[4] = (lane == 0) ? l : -3.0e38f;
        }
        float m = L[0];
        #pragma unroll
        for (int i = 1; i < 5; ++i) m = fmaxf(m, L[i]);
        #pragma unroll
        for (int off = 16; off > 0; off >>= 1) m = fmaxf(m, __shfl_xor(m, off, 32));
        float e[5], sum = 0.0f;
        #pragma unroll
        for (int i = 0; i < 5; ++i) { e[i] = __expf(L[i] - m); sum += e[i]; }
        #pragma unroll
        for (int off = 16; off > 0; off >>= 1) sum += __shfl_xor(sum, off, 32);
        const float inv = 1.0f / sum;
        #pragma unroll
        for (int i = 0; i < 4; ++i) wS[wave][lane + i * 32] = e[i] * inv;
        if (lane == 0) wS[wave][128] = e[4] * inv;
    }

    // ---- V = v@Wv, then ctx = w . V (+ folded wv_b / bias_v terms) ---------
    __syncthreads();
    loadA(gv);
    __syncthreads();

    v8f accV[8][2];
    #pragma unroll
    for (int mt = 0; mt < 8; ++mt) { accV[mt][0] = zero8(); accV[mt][1] = zero8(); }
    for (int kk = 0; kk < 8; ++kk) {
        const int kb = kk * 32;
        v16bf b0 = load_bfrag(WvT, n0,      kb, lane);
        v16bf b1 = load_bfrag(WvT, n0 + 16, kb, lane);
        #pragma unroll
        for (int mt = 0; mt < 8; ++mt) {
            v16bf a = load_afrag(&aBuf[0][0], mt * 16, kb, lane);
            accV[mt][0] = __builtin_amdgcn_wmma_f32_16x16x32_bf16(false, a, false, b0,
                              (short)0, accV[mt][0], false, false);
            accV[mt][1] = __builtin_amdgcn_wmma_f32_16x16x32_bf16(false, a, false, b1,
                              (short)0, accV[mt][1], false, false);
        }
    }
    {
        float c0 = 0.0f, c1 = 0.0f;
        #pragma unroll
        for (int mt = 0; mt < 8; ++mt) {
            int rowBase = mt * 16 + ((lane & 16) ? 8 : 0);
            #pragma unroll
            for (int r = 0; r < 8; ++r) {
                float w = wS[wave][rowBase + r];
                c0 += accV[mt][0][r] * w;
                c1 += accV[mt][1][r] * w;
            }
        }
        c0 += __shfl_xor(c0, 16, 32);
        c1 += __shfl_xor(c1, 16, 32);
        const float w128 = wS[wave][128];
        const float wsum = 1.0f - w128;   // softmax weights sum to 1
        if (lane < 16) {
            int col0 = n0 + lane;
            int col1 = col0 + 16;
            ctxS[col0] = c0 + wsum * wv_b[col0] + w128 * bias_v[col0];
            ctxS[col1] = c1 + wsum * wv_b[col1] + w128 * bias_v[col1];
        }
    }
    __syncthreads();

    // ---- out = ctx @ Wout + out_b (1x256, VALU) ----------------------------
    {
        float acc = out_b[tid];
        const bf16_t* wrow = WoT + tid * DDIM;
        #pragma unroll 8
        for (int kk = 0; kk < DDIM; ++kk) acc += ctxS[kk] * (float)wrow[kk];
        out[b * DDIM + tid] = acc;
    }
}

extern "C" void kernel_launch(void* const* d_in, const int* in_sizes, int n_in,
                              void* d_out, int out_size, void* d_ws, size_t ws_size,
                              hipStream_t stream) {
    (void)in_sizes; (void)n_in; (void)out_size; (void)ws_size;
    const float* q       = (const float*)d_in[0];
    const float* k       = (const float*)d_in[1];
    const float* v       = (const float*)d_in[2];
    const float* pw      = (const float*)d_in[3];
    const int*   mask    = (const int*)  d_in[4];
    const float* wq_w    = (const float*)d_in[5];
    const float* wq_b    = (const float*)d_in[6];
    const float* wk_w    = (const float*)d_in[7];
    const float* wk_b    = (const float*)d_in[8];
    const float* wv_w    = (const float*)d_in[9];
    const float* wv_b    = (const float*)d_in[10];
    const float* wp_w    = (const float*)d_in[11];
    const float* wp_b    = (const float*)d_in[12];
    const float* out_w   = (const float*)d_in[13];
    const float* out_b   = (const float*)d_in[14];
    const float* aggv    = (const float*)d_in[15];
    const float* posb    = (const float*)d_in[16];
    const float* bias_k  = (const float*)d_in[17];
    const float* bias_v  = (const float*)d_in[18];
    const float* bias_pb = (const float*)d_in[19];
    const float* bias_pw = (const float*)d_in[20];

    bf16_t* wt = (bf16_t*)d_ws;   // 5 * 256*256 bf16 = 640 KB

    prep_weights<<<(5 * DDIM * DDIM) / 256, 256, 0, stream>>>(wq_w, wk_w, wp_w, wv_w,
                                                              out_w, wt);
    mha_fused<<<BDIM, 256, 0, stream>>>(q, k, v, pw, mask, wt,
                                        wq_b, wk_b, wv_b, wp_b, out_b,
                                        aggv, posb, bias_k, bias_v, bias_pb, bias_pw,
                                        (float*)d_out);
}
